// OldQuantLinear_54331336294693
// MI455X (gfx1250) — matrix operand order repaired and checked by reading, not compiled, not run on hardware
//
#include <hip/hip_runtime.h>
#include <hip/hip_bf16.h>

typedef _Float16 v4h __attribute__((ext_vector_type(4)));
typedef _Float16 v8h __attribute__((ext_vector_type(8)));
typedef _Float16 v16h __attribute__((ext_vector_type(16)));
typedef float    v8f __attribute__((ext_vector_type(8)));

#define KDIM 4096
#define NDIM 4096

// ---------------------------------------------------------------------------
// Prep: convert x (fp32) -> f16 copy in workspace, and rowsum[m] = sum_k f16(x)
// (summing the *converted* values so the 1024-bias term cancels exactly against
//  what the WMMA accumulates).
// ---------------------------------------------------------------------------
__global__ void __launch_bounds__(256) gptq_prep_kernel(
    const float* __restrict__ x, _Float16* __restrict__ x16,
    float* __restrict__ rowsum)
{
    const int row = blockIdx.x;
    const int tid = threadIdx.x;
    const float*    xr = x   + (size_t)row * KDIM;
    _Float16*       xo = x16 + (size_t)row * KDIM;

    float s = 0.0f;
    #pragma unroll
    for (int k = tid * 4; k < KDIM; k += 256 * 4) {
        float4 v = *(const float4*)(xr + k);
        v4h h;
        h.x = (_Float16)v.x; h.y = (_Float16)v.y;
        h.z = (_Float16)v.z; h.w = (_Float16)v.w;
        *(v4h*)(xo + k) = h;
        s += (float)h.x + (float)h.y + (float)h.z + (float)h.w;
    }

    __shared__ float red[256];
    red[tid] = s;
    __syncthreads();
    #pragma unroll
    for (int off = 128; off > 0; off >>= 1) {
        if (tid < off) red[tid] += red[tid + off];
        __syncthreads();
    }
    if (tid == 0) rowsum[row] = red[0];
}

// ---------------------------------------------------------------------------
// Unpack one packed int32 (8 nibbles, LSB-first along K) into 8 f16 values
// biased by +1024 via exponent splice:  0x6400 | nib == f16(1024 + nib).
// ---------------------------------------------------------------------------
union V16U { v16h v; v8h h[2]; unsigned d[8]; };

__device__ __forceinline__ void unpack8(unsigned w, unsigned* d4) {
    #pragma unroll
    for (int t = 0; t < 4; ++t) {
        unsigned b = w >> (8 * t);               // nib(2t) in [3:0], nib(2t+1) in [7:4]
        d4[t] = (b & 0x0000000Fu)                // low  half: 1024 + nib(2t)
              | ((b & 0x000000F0u) << 12)        // high half: 1024 + nib(2t+1)
              | 0x64006400u;
    }
}

// ---------------------------------------------------------------------------
// WMMA GEMM: out[M,N] = f16(x)[M,K] @ (1024+q)[K,N], epilogue applies
//   y = s[n]*acc - (1024*s[n] + z[n])*rowsum[m] + bias[n]
// Block: 256 thr = 8 waves (2x4). Wave: 64x64 (4x4 of 16x16x32 WMMA tiles).
// Block tile: 128x256. Grid: (N/256, M/128).
// ---------------------------------------------------------------------------
__global__ void __launch_bounds__(256) gptq_wmma_kernel(
    const _Float16* __restrict__ x16, const unsigned* __restrict__ qw,
    const float* __restrict__ scales, const float* __restrict__ zeros,
    const float* __restrict__ bias,   const float* __restrict__ rowsum,
    float* __restrict__ out)
{
    const int lane = threadIdx.x & 31;
    const int wave = threadIdx.x >> 5;
    const int wm   = wave >> 2;          // 0..1
    const int wn   = wave & 3;           // 0..3
    const int mBase = blockIdx.y * 128 + wm * 64;
    const int nBase = blockIdx.x * 256 + wn * 64;
    const int ml = lane & 15;
    const int hi = lane >> 4;            // lane half

    v8f acc[4][4];
    #pragma unroll
    for (int i = 0; i < 4; ++i)
        #pragma unroll
        for (int j = 0; j < 4; ++j)
            acc[i][j] = (v8f){0.f,0.f,0.f,0.f,0.f,0.f,0.f,0.f};

    // A base pointers: row m = mBase + i*16 + ml ; per-lane K chunks at
    // k0 + hi*16 (elements 0..7) and k0 + hi*16 + 8 (elements 8..15).
    const _Float16* aPtr[4];
    #pragma unroll
    for (int i = 0; i < 4; ++i)
        aPtr[i] = x16 + (size_t)(mBase + i * 16 + ml) * KDIM + hi * 16;

    // B: column n = nBase + j*16 + ml ; lane half covers K = k0 + hi*16 + 0..15
    // => packed rows (k0>>3) + 2*hi and +1.
    const unsigned* bPtr[4];
    #pragma unroll
    for (int j = 0; j < 4; ++j)
        bPtr[j] = qw + (size_t)(2 * hi) * NDIM + (nBase + j * 16 + ml);

    for (int k0 = 0; k0 < KDIM; k0 += 32) {
        v16h A[4], B[4];

        #pragma unroll
        for (int i = 0; i < 4; ++i) {
            V16U u;
            u.h[0] = *(const v8h*)(aPtr[i] + k0);        // K = k0+hi*16 .. +7
            u.h[1] = *(const v8h*)(aPtr[i] + k0 + 8);    // K = k0+hi*16+8 .. +15
            A[i] = u.v;
        }

        const size_t qrow = (size_t)(k0 >> 3) * NDIM;    // +2*hi folded into bPtr
        #pragma unroll
        for (int j = 0; j < 4; ++j) {
            unsigned w0 = bPtr[j][qrow];                 // K = k0+hi*16 .. +7
            unsigned w1 = bPtr[j][qrow + NDIM];          // K = k0+hi*16+8 .. +15
            V16U u;
            unpack8(w0, &u.d[0]);
            unpack8(w1, &u.d[4]);
            B[j] = u.v;
        }

        #pragma unroll
        for (int i = 0; i < 4; ++i)
            #pragma unroll
            for (int j = 0; j < 4; ++j)
                acc[i][j] = __builtin_amdgcn_wmma_f32_16x16x32_f16(
                    false, A[i], false, B[j], (short)0, acc[i][j], false, false);
    }

    // Epilogue. C/D layout: lane -> col n = ml, element p -> row = p + hi*8.
    #pragma unroll
    for (int j = 0; j < 4; ++j) {
        const int n = nBase + j * 16 + ml;
        const float sc = scales[n];
        const float coef = -(1024.0f * sc + zeros[n]);   // cancels +1024 bias, applies zero
        const float bs = bias[n];
        #pragma unroll
        for (int i = 0; i < 4; ++i) {
            const int rbase = mBase + i * 16 + hi * 8;
            #pragma unroll
            for (int p = 0; p < 8; ++p) {
                const int row = rbase + p;
                out[(size_t)row * NDIM + n] = sc * acc[i][j][p] + coef * rowsum[row] + bs;
            }
        }
    }
}

extern "C" void kernel_launch(void* const* d_in, const int* in_sizes, int n_in,
                              void* d_out, int out_size, void* d_ws, size_t ws_size,
                              hipStream_t stream) {
    const float*    x  = (const float*)d_in[0];
    const unsigned* qw = (const unsigned*)d_in[1];
    const float*    sc = (const float*)d_in[2];
    const float*    zr = (const float*)d_in[3];
    const float*    bs = (const float*)d_in[4];
    float*          out = (float*)d_out;

    const int M = in_sizes[0] / KDIM;            // 8192

    _Float16* x16    = (_Float16*)d_ws;
    float*    rowsum = (float*)((char*)d_ws + (size_t)M * KDIM * sizeof(_Float16));

    gptq_prep_kernel<<<M, 256, 0, stream>>>(x, x16, rowsum);

    dim3 grid(NDIM / 256, M / 128);
    gptq_wmma_kernel<<<grid, 256, 0, stream>>>(x16, qw, sc, zr, bs, rowsum, out);
}
